// PSI_42640435314810
// MI455X (gfx1250) — matrix operand order repaired and checked by reading, not compiled, not run on hardware
//
#include <hip/hip_runtime.h>
#include <math.h>

typedef __bf16 bf16;
typedef __bf16 v16bf __attribute__((ext_vector_type(16)));
typedef float  v8f   __attribute__((ext_vector_type(8)));

#define BDIM   4
#define SDIM   4096
#define DDIM   1024
#define NCHUNK 32
#define CLEN   (SDIM / NCHUNK)   // 128
#define BSROWS (BDIM * SDIM)     // 16384

union Frag16 { v16bf v; uint4 u[2]; };

// ---------------------------------------------------------------- fp32 -> bf16
__global__ void cvt_f32_bf16(const float* __restrict__ src, bf16* __restrict__ dst, long n) {
    long i = (long)blockIdx.x * blockDim.x + threadIdx.x;
    long stride = (long)gridDim.x * blockDim.x;
    for (; i < n; i += stride) dst[i] = (bf16)src[i];
}

// ------------------------------------------------- scan pass 1: chunk sums of omega_scaled
__global__ void chunk_sum_omega(const float* __restrict__ omega,
                                const float* __restrict__ log_scale,
                                float* __restrict__ chunkO) {
    int tid = blockIdx.x * blockDim.x + threadIdx.x;   // B*NCHUNK*D threads
    int d = tid % DDIM;
    int c = (tid / DDIM) % NCHUNK;
    int b = tid / (DDIM * NCHUNK);
    float es = __expf(log_scale[d]);
    float sum = 0.f;
    long base = ((long)b * SDIM + (long)c * CLEN) * DDIM + d;
    for (int j = 0; j < CLEN; ++j) {
        int s = c * CLEN + j;
        sum += omega[base + (long)j * DDIM] * es * rsqrtf((float)(s + 1));
    }
    chunkO[tid] = sum;
}

// ------------------------------------------------- tiny exclusive scan over the 32 chunks
__global__ void excl_scan_chunks(float* __restrict__ buf) {
    int tid = blockIdx.x * blockDim.x + threadIdx.x;   // B*D threads
    int d = tid % DDIM;
    int b = tid / DDIM;
    float run = 0.f;
    for (int c = 0; c < NCHUNK; ++c) {
        long idx = ((long)b * NCHUNK + c) * DDIM + d;
        float t = buf[idx];
        buf[idx] = run;
        run += t;
    }
}

// ------------------------------------------------- scan pass 2: phi, cos/sin, cm_{real,imag}
__global__ void scan_phase1(const float* __restrict__ omega,
                            const float* __restrict__ x,
                            const float* __restrict__ log_scale,
                            const float* __restrict__ chunkO,
                            float* __restrict__ phiBuf,
                            bf16*  __restrict__ ctx,
                            float* __restrict__ chunkR,
                            float* __restrict__ chunkI) {
    int tid = blockIdx.x * blockDim.x + threadIdx.x;   // B*NCHUNK*D threads
    int d = tid % DDIM;
    int c = (tid / DDIM) % NCHUNK;
    int b = tid / (DDIM * NCHUNK);
    float es = __expf(log_scale[d]);
    float phi = chunkO[tid];                           // exclusive base for this chunk
    float sr = 0.f, si = 0.f;
    for (int j = 0; j < CLEN; ++j) {
        int s = c * CLEN + j;
        long ix = ((long)b * SDIM + s) * DDIM + d;
        phi += omega[ix] * es * rsqrtf((float)(s + 1));
        phiBuf[ix] = phi;
        float sv, cv;
        __sincosf(phi, &sv, &cv);
        float xv = x[ix];
        float cr = xv * cv;
        float ci = xv * sv;
        long cbase = ((long)b * SDIM + s) * (4 * DDIM) + d;
        ctx[cbase]         = (bf16)cr;
        ctx[cbase + DDIM]  = (bf16)ci;
        sr += cr;
        si += ci;
    }
    chunkR[tid] = sr;
    chunkI[tid] = si;
}

// ------------------------------------------------- scan pass 3: mem, retrieved components
__global__ void scan_phase2(const float* __restrict__ phiBuf,
                            bf16*  __restrict__ ctx,
                            const float* __restrict__ chunkR,
                            const float* __restrict__ chunkI) {
    int tid = blockIdx.x * blockDim.x + threadIdx.x;
    int d = tid % DDIM;
    int c = (tid / DDIM) % NCHUNK;
    int b = tid / (DDIM * NCHUNK);
    float rr = chunkR[tid];   // exclusive bases
    float ri = chunkI[tid];
    for (int j = 0; j < CLEN; ++j) {
        int s = c * CLEN + j;
        long ix = ((long)b * SDIM + s) * DDIM + d;
        long cbase = ((long)b * SDIM + s) * (4 * DDIM) + d;
        float cr = (float)ctx[cbase];
        float ci = (float)ctx[cbase + DDIM];
        rr += cr;
        ri += ci;
        float inv = 1.f / (float)(s + 1);
        float mr = rr * inv;
        float mi = ri * inv;
        float sv, cv;
        __sincosf(phiBuf[ix], &sv, &cv);
        ctx[cbase + 2 * DDIM] = (bf16)(mr * cv + mi * sv);
        ctx[cbase + 3 * DDIM] = (bf16)(mi * cv - mr * sv);
    }
}

// ------------------------------------------------- LayerNorm over 4096, in place on bf16 ctx
__global__ __launch_bounds__(256)
void layernorm_inplace(bf16* __restrict__ ctx,
                       const float* __restrict__ gamma,
                       const float* __restrict__ beta) {
    __shared__ float red[256];
    long row = blockIdx.x;
    bf16* p = ctx + row * (long)(4 * DDIM);
    int t = threadIdx.x;
    float vals[16];
    float sum = 0.f;
    #pragma unroll
    for (int k = 0; k < 16; ++k) {
        vals[k] = (float)p[t + 256 * k];
        sum += vals[k];
    }
    red[t] = sum;
    __syncthreads();
    for (int o = 128; o > 0; o >>= 1) {
        if (t < o) red[t] += red[t + o];
        __syncthreads();
    }
    float mean = red[0] * (1.f / 4096.f);
    __syncthreads();
    float sq = 0.f;
    #pragma unroll
    for (int k = 0; k < 16; ++k) {
        float dd = vals[k] - mean;
        sq += dd * dd;
    }
    red[t] = sq;
    __syncthreads();
    for (int o = 128; o > 0; o >>= 1) {
        if (t < o) red[t] += red[t + o];
        __syncthreads();
    }
    float var  = red[0] * (1.f / 4096.f);
    float rstd = rsqrtf(var + 1e-5f);
    #pragma unroll
    for (int k = 0; k < 16; ++k) {
        int ci = t + 256 * k;
        p[ci] = (bf16)((vals[k] - mean) * rstd * gamma[ci] + beta[ci]);
    }
}

// ------------------------------------------------- async global -> LDS helpers (CDNA5)
__device__ __forceinline__ void async_copy16(const bf16* g, bf16* l) {
    unsigned ldsOff = (unsigned)(uintptr_t)(void*)l;      // low 32 bits = LDS offset
    unsigned long long ga = (unsigned long long)(const void*)g;
    asm volatile("global_load_async_to_lds_b128 %0, %1, off"
                 :: "v"(ldsOff), "v"(ga) : "memory");
}
#define WAIT_ASYNC(n) asm volatile("s_wait_asynccnt " #n ::: "memory")

// ------------------------------------------------- WMMA GEMM: C[M,N] = A[M,K] * W[N,K]^T + bias
// epi: 0 -> f32 out, 1 -> exact GELU -> bf16 out, 2 -> +resid -> f32 out
// 8 waves as 4x2, wave tile 64x64 (16 accumulators) -> 1 ds_load_b128 per WMMA.
#define BM  256
#define BN  128
#define BK  64
#define LDW (BK + 8)   // 72 bf16 = 144B row stride: 16B aligned, conflict-breaking

__global__ __launch_bounds__(256)
void wmma_gemm(const bf16* __restrict__ A, const bf16* __restrict__ Bm,
               const float* __restrict__ bias,
               float* __restrict__ outF, bf16* __restrict__ outB,
               const float* __restrict__ resid,
               int M, int N, int K, int epi) {
    __shared__ __align__(16) bf16 ldsA[2][BM * LDW];   // 2 x 36 KB
    __shared__ __align__(16) bf16 ldsB[2][BN * LDW];   // 2 x 18 KB

    int t = threadIdx.x;
    int lane = t & 31, wid = t >> 5;
    int wm = wid & 3, wn = wid >> 2;        // 4x2 wave grid -> 256x128 block tile
    int m0 = blockIdx.y * BM;
    int n0 = blockIdx.x * BN;
    int laneHi = lane >> 4;
    int lane16 = lane & 15;

    v8f acc[4][4];
    #pragma unroll
    for (int i = 0; i < 4; ++i)
        #pragma unroll
        for (int j = 0; j < 4; ++j)
            #pragma unroll
            for (int e = 0; e < 8; ++e) acc[i][j][e] = 0.f;

    int lrow = t >> 3;          // 0..31
    int lcol = (t & 7) * 8;     // 0..56, 16B granules

    const int nk = K / BK;

    // Issue one K-tile of async global->LDS copies (12 b128 ops per thread).
    auto issueTile = [&](int kt, int buf) {
        long k0 = (long)kt * BK;
        #pragma unroll
        for (int rr = 0; rr < BM; rr += 32)
            async_copy16(A + (long)(m0 + lrow + rr) * K + k0 + lcol,
                         &ldsA[buf][(lrow + rr) * LDW + lcol]);
        #pragma unroll
        for (int rr = 0; rr < BN; rr += 32)
            async_copy16(Bm + (long)(n0 + lrow + rr) * K + k0 + lcol,
                         &ldsB[buf][(lrow + rr) * LDW + lcol]);
    };

    issueTile(0, 0);

    for (int kt = 0; kt < nk; ++kt) {
        int cur = kt & 1;
        if (kt + 1 < nk) {
            issueTile(kt + 1, cur ^ 1);   // 12 more in flight
            WAIT_ASYNC(12);               // current tile's 12 have landed
        } else {
            WAIT_ASYNC(0);
        }
        __syncthreads();

        #pragma unroll
        for (int kk = 0; kk < BK; kk += 32) {
            Frag16 afr[4], bfr[4];
            #pragma unroll
            for (int i = 0; i < 4; ++i) {
                int row = wm * 64 + 16 * i + lane16;
                int kb = kk + 8 * laneHi;            // A: K in {8h..8h+7} U {8h+16..8h+23}
                afr[i].u[0] = *(const uint4*)(&ldsA[cur][row * LDW + kb]);
                afr[i].u[1] = *(const uint4*)(&ldsA[cur][row * LDW + kb + 16]);
            }
            #pragma unroll
            for (int j = 0; j < 4; ++j) {
                int col = wn * 64 + 16 * j + lane16;
                int kb = kk + 8 * laneHi;            // B: K in {8h..8h+7} U {8h+16..8h+23}
                bfr[j].u[0] = *(const uint4*)(&ldsB[cur][col * LDW + kb]);
                bfr[j].u[1] = *(const uint4*)(&ldsB[cur][col * LDW + kb + 16]);
            }
            #pragma unroll
            for (int i = 0; i < 4; ++i)
                #pragma unroll
                for (int j = 0; j < 4; ++j)
                    acc[i][j] = __builtin_amdgcn_wmma_f32_16x16x32_bf16(
                        false, afr[i].v, false, bfr[j].v,
                        (short)0, acc[i][j], false, false);
        }
        __syncthreads();   // protect the buffer targeted by the next issue
    }

    #pragma unroll
    for (int i = 0; i < 4; ++i) {
        #pragma unroll
        for (int j = 0; j < 4; ++j) {
            int col = n0 + wn * 64 + 16 * j + lane16;
            float bv = bias[col];
            #pragma unroll
            for (int r = 0; r < 8; ++r) {
                int row = m0 + wm * 64 + 16 * i + 8 * laneHi + r;
                float v = acc[i][j][r] + bv;
                long idx = (long)row * N + col;
                if (epi == 1) {
                    v = 0.5f * v * (1.0f + erff(v * 0.70710678118654752f));
                    outB[idx] = (bf16)v;
                } else if (epi == 2) {
                    outF[idx] = v + resid[idx];
                } else {
                    outF[idx] = v;
                }
            }
        }
    }
}

// ----------------------------------------------------------------------------
extern "C" void kernel_launch(void* const* d_in, const int* in_sizes, int n_in,
                              void* d_out, int out_size, void* d_ws, size_t ws_size,
                              hipStream_t stream) {
    const float* x         = (const float*)d_in[0];
    const float* W_omega   = (const float*)d_in[1];
    const float* b_omega   = (const float*)d_in[2];
    const float* log_scale = (const float*)d_in[3];
    const float* ln_gamma  = (const float*)d_in[4];
    const float* ln_beta   = (const float*)d_in[5];
    const float* W1        = (const float*)d_in[6];
    const float* b1        = (const float*)d_in[7];
    const float* W2        = (const float*)d_in[8];
    const float* b2        = (const float*)d_in[9];
    float* out = (float*)d_out;

    char* ws = (char*)d_ws;
    size_t off = 0;
    auto alloc = [&](size_t bytes) -> void* {
        void* p = ws + off;
        off = (off + bytes + 255) & ~(size_t)255;
        return p;
    };

    const long nXD  = (long)BSROWS * DDIM;           // 16.7M
    bf16*  xbf    = (bf16*)alloc(nXD * 2);
    bf16*  wobf   = (bf16*)alloc((size_t)DDIM * DDIM * 2);
    bf16*  w1bf   = (bf16*)alloc((size_t)2 * DDIM * 4 * DDIM * 2);
    bf16*  w2bf   = (bf16*)alloc((size_t)DDIM * 2 * DDIM * 2);
    float* omega  = (float*)alloc(nXD * 4);
    float* phiBuf = (float*)alloc(nXD * 4);
    bf16*  ctx    = (bf16*)alloc((size_t)BSROWS * 4 * DDIM * 2);
    bf16*  h1bf   = (bf16*)alloc((size_t)BSROWS * 2 * DDIM * 2);
    float* chunkO = (float*)alloc((size_t)BDIM * NCHUNK * DDIM * 4);
    float* chunkR = (float*)alloc((size_t)BDIM * NCHUNK * DDIM * 4);
    float* chunkI = (float*)alloc((size_t)BDIM * NCHUNK * DDIM * 4);

    // --- bf16 copies of operands
    cvt_f32_bf16<<<8192, 256, 0, stream>>>(x, xbf, nXD);
    cvt_f32_bf16<<<2048, 256, 0, stream>>>(W_omega, wobf, (long)DDIM * DDIM);
    cvt_f32_bf16<<<8192, 256, 0, stream>>>(W1, w1bf, (long)2 * DDIM * 4 * DDIM);
    cvt_f32_bf16<<<4096, 256, 0, stream>>>(W2, w2bf, (long)DDIM * 2 * DDIM);

    // --- G1: omega = x @ W_omega^T + b_omega
    dim3 g1(DDIM / BN, BSROWS / BM);
    wmma_gemm<<<g1, 256, 0, stream>>>(xbf, wobf, b_omega, omega, nullptr, nullptr,
                                      BSROWS, DDIM, DDIM, 0);

    // --- chunked scans
    int scanBlocks = (BDIM * NCHUNK * DDIM) / 256;   // 512
    chunk_sum_omega<<<scanBlocks, 256, 0, stream>>>(omega, log_scale, chunkO);
    excl_scan_chunks<<<(BDIM * DDIM) / 256, 256, 0, stream>>>(chunkO);
    scan_phase1<<<scanBlocks, 256, 0, stream>>>(omega, x, log_scale, chunkO,
                                                phiBuf, ctx, chunkR, chunkI);
    excl_scan_chunks<<<(BDIM * DDIM) / 256, 256, 0, stream>>>(chunkR);
    excl_scan_chunks<<<(BDIM * DDIM) / 256, 256, 0, stream>>>(chunkI);
    scan_phase2<<<scanBlocks, 256, 0, stream>>>(phiBuf, ctx, chunkR, chunkI);

    // --- LayerNorm (in place on bf16 context)
    layernorm_inplace<<<BSROWS, 256, 0, stream>>>(ctx, ln_gamma, ln_beta);

    // --- G2: h1 = GELU(ctx @ W1^T + b1)  (bf16 out)
    dim3 g2((2 * DDIM) / BN, BSROWS / BM);
    wmma_gemm<<<g2, 256, 0, stream>>>(ctx, w1bf, b1, nullptr, h1bf, nullptr,
                                      BSROWS, 2 * DDIM, 4 * DDIM, 1);

    // --- G3: out = x + h1 @ W2^T + b2  (f32 out, fused residual)
    dim3 g3(DDIM / BN, BSROWS / BM);
    wmma_gemm<<<g3, 256, 0, stream>>>(h1bf, w2bf, b2, out, nullptr, x,
                                      BSROWS, DDIM, 2 * DDIM, 2);
}
